// VICRegLLoss_10977936408943
// MI455X (gfx1250) — compile-verified, matrix operand order checked by reading, not run on hardware
//
#include <hip/hip_runtime.h>
#include <math.h>

// ---------------- constants from the reference ----------------
#define LAMBDA_ 25.0f
#define MU_     25.0f
#define NU_     1.0f
#define ALPHA_  0.75f
#define EPS_    1e-4f

static constexpr int Bn = 256;     // batch
static constexpr int DG = 2048;    // global feature dim
static constexpr int DL = 512;     // local feature dim
static constexpr int NG = 14 * 14; // 196
static constexpr int NL = 7 * 7;   // 49
static constexpr int KA = 20;      // num_matches a-direction
static constexpr int KB = 4;       // num_matches b-direction

typedef float v2f  __attribute__((ext_vector_type(2)));
typedef float v8f  __attribute__((ext_vector_type(8)));
typedef _Float16 v16h __attribute__((ext_vector_type(16)));
typedef unsigned int u32x4 __attribute__((ext_vector_type(4)));
typedef int i32x4v __attribute__((ext_vector_type(4)));
typedef int i32x8v __attribute__((ext_vector_type(8)));

#if __has_builtin(__builtin_amdgcn_tensor_load_to_lds) && __has_builtin(__builtin_amdgcn_s_wait_tensorcnt)
#define HAVE_TDM 1
#else
#define HAVE_TDM 0
#endif

// =====================================================================
// Wave-level 16x16 Gram tile accumulate: acc += sum_k A(i,k)*B(j,k)
//   A(i,k) = pA[i*sAi + k*sAk],  B(j,k) = pB[j*sBj + k*sBk]
// Row/col indices clamped (callers skip OOB writes). K multiple of 32.
// Primary: V_WMMA_F32_16X16X4_F32; fallback: V_WMMA_F32_16X16X32_F16.
// =====================================================================
__device__ __forceinline__ v8f gram_tile(v8f acc,
                                         const float* __restrict__ pA, long sAi, long sAk,
                                         const float* __restrict__ pB, long sBj, long sBk,
                                         int K, int i0, int j0, int lane,
                                         int maxI, int maxJ, bool pf) {
  const int mi = lane & 15;
  int i = i0 + mi; if (i > maxI - 1) i = maxI - 1;
  int j = j0 + mi; if (j > maxJ - 1) j = maxJ - 1;
  const float* ap = pA + (long)i * sAi;
  const float* bp = pB + (long)j * sBj;
#if __has_builtin(__builtin_amdgcn_wmma_f32_16x16x4_f32)
  // A 16x4 fp32: lanes 0-15 -> K = k+0,k+1 ; lanes 16-31 -> K = k+2,k+3
  const int koff = (lane >> 4) << 1;
  for (int k = 0; k < K; k += 4) {
    if (pf && (k & 63) == 0 && k + 64 < K) {
      __builtin_prefetch(ap + (long)(k + 64 + koff) * sAk, 0, 3);
      __builtin_prefetch(bp + (long)(k + 64 + koff) * sBk, 0, 3);
    }
    v2f av, bv;
    av.x = ap[(long)(k + koff)     * sAk];
    av.y = ap[(long)(k + koff + 1) * sAk];
    bv.x = bp[(long)(k + koff)     * sBk];
    bv.y = bp[(long)(k + koff + 1) * sBk];
    acc = __builtin_amdgcn_wmma_f32_16x16x4_f32(false, av, false, bv,
                                                (short)0, acc, false, false);
  }
#else
  const int hi8 = (lane >> 4) * 8;
  for (int k = 0; k < K; k += 32) {
    v16h av, bv;
#pragma unroll
    for (int jj = 0; jj < 8; ++jj) {
      int kb = ((jj < 4) ? 0 : 16) + hi8 + 2 * (jj & 3);
      av[2 * jj]     = (_Float16)ap[(long)(k + kb)     * sAk];
      av[2 * jj + 1] = (_Float16)ap[(long)(k + kb + 1) * sAk];
      bv[2 * jj]     = (_Float16)bp[(long)(k + kb)     * sBk];
      bv[2 * jj + 1] = (_Float16)bp[(long)(k + kb + 1) * sBk];
    }
    acc = __builtin_amdgcn_wmma_f32_16x16x32_f16(false, av, false, bv,
                                                 (short)0, acc, false, false);
  }
#endif
  return acc;
}

#if HAVE_TDM
// Issue one TDM 2-D tile load: global (tileH rows x tileW fp32 cols, row pitch
// rowStrideElems) -> LDS at byte offset ldsOff (rows packed contiguously).
// D# bit layout per CDNA5 ISA 8.3/8.4; tensor dims == tile dims (tiles are
// always fully in-bounds here).
__device__ __forceinline__ void tdm_load_2d(unsigned ldsOff, const float* gaddr,
                                            int tileW, int tileH, long rowStrideElems) {
  unsigned long long ga = (unsigned long long)(uintptr_t)gaddr;
  u32x4 g0;
  g0[0] = 1u;                                   // count=1, user descriptor
  g0[1] = ldsOff;                               // lds_addr (bytes)
  g0[2] = (unsigned)(ga & 0xffffffffu);         // global_addr[31:0]
  g0[3] = (unsigned)((ga >> 32) & 0x01ffffffu)  // global_addr[56:32]
          | (2u << 30);                         // type=2 ("image")
  i32x8v g1;
  g1[0] = 0x20000;                              // data_size=2 (4 bytes)
  g1[1] = (int)(((unsigned)tileW & 0xffffu) << 16);           // tensor_dim0 lo16
  g1[2] = (int)(((unsigned)tileH << 16) | ((unsigned)tileW >> 16)); // dim0 hi / dim1 lo
  g1[3] = (int)(((unsigned)tileW & 0xffffu) << 16);           // tile_dim0
  g1[4] = tileH;                                // tile_dim1 (tile_dim2 = 0)
  g1[5] = (int)(unsigned)(rowStrideElems & 0xffffffffL);      // dim0_stride lo32
  g1[6] = (int)(unsigned)((rowStrideElems >> 32) & 0xffffL);  // dim0_stride hi16
  g1[7] = 0;
  i32x4v z4 = {0, 0, 0, 0};
#if defined(__clang_major__) && __clang_major__ >= 23
  i32x8v z8 = {0, 0, 0, 0, 0, 0, 0, 0};
  __builtin_amdgcn_tensor_load_to_lds(g0, g1, z4, z4, z8, 0);
#else
  __builtin_amdgcn_tensor_load_to_lds(g0, g1, z4, z4, 0);
#endif
}
#endif

// ---------------- small utility kernels ----------------
__global__ void k_init(float* acc) {
  if (threadIdx.x < 16) acc[threadIdx.x] = 0.0f;
}

__global__ void k_finalize(const float* acc, float* out) { out[0] = acc[0]; }

__global__ void k_sumsq_diff(const float* __restrict__ x, const float* __restrict__ y,
                             long n, float scale, float* acc) {
  __shared__ float red[256];
  long i = (long)blockIdx.x * blockDim.x + threadIdx.x;
  long stride = (long)gridDim.x * blockDim.x;
  float s = 0.0f;
  for (; i < n; i += stride) { float d = x[i] - y[i]; s += d * d; }
  red[threadIdx.x] = s;
  __syncthreads();
  for (int t = 128; t; t >>= 1) {
    if ((int)threadIdx.x < t) red[threadIdx.x] += red[threadIdx.x + t];
    __syncthreads();
  }
  if (threadIdx.x == 0) atomicAdd(acc, scale * red[0]);
}

// Per-column mean + variance-loss contribution (ddof=1).
__global__ void k_colstats(const float* __restrict__ x, int B, int D, long rowStride,
                           float* __restrict__ means, float varScale, float* acc) {
  __shared__ float red[256];
  int d = blockIdx.x * blockDim.x + threadIdx.x;
  float contrib = 0.0f;
  if (d < D) {
    float s = 0.0f, ss = 0.0f;
    for (int k = 0; k < B; ++k) {
      float v = x[(long)k * rowStride + d];
      s += v; ss += v * v;
    }
    float mean = s / (float)B;
    means[d] = mean;
    float var = (ss - (float)B * mean * mean) / (float)(B - 1);
    float sd = sqrtf(var + EPS_);
    contrib = fmaxf(1.0f - sd, 0.0f);
  }
  red[threadIdx.x] = contrib;
  __syncthreads();
  for (int t = 128; t; t >>= 1) {
    if ((int)threadIdx.x < t) red[threadIdx.x] += red[threadIdx.x + t];
    __syncthreads();
  }
  if (threadIdx.x == 0) atomicAdd(acc, varScale * red[0]);
}

// =====================================================================
// Covariance off-diagonal sum-of-squares.
// One block (8 waves) owns a 64x32 region of the DxD Gram (4x2 16x16 tiles).
// TDM path: wave 0 DMAs a 32x64 A-panel + 32x32 B-panel into LDS per K-chunk
// (s_wait_tensorcnt + workgroup barrier), all waves run WMMA from LDS.
// gridDim.x = (D/64)*(D/32); gridDim.y iterates positions (pos = y*posStride).
// =====================================================================
__global__ void k_cov_wmma(const float* __restrict__ x, int B, int D, long rowStride,
                           long posStride, const float* __restrict__ means,
                           float covScale, float* acc) {
  const int lane = threadIdx.x & 31;
  const int wave = threadIdx.x >> 5;
  const int nRegD = D >> 5;
  const int c0 = (blockIdx.x / nRegD) << 6;   // 64-wide row block
  const int d0 = (blockIdx.x % nRegD) << 5;   // 32-wide col block
  const long pos = (long)blockIdx.y * posStride;
  const float* xp = x + pos;
  const float* mp = means + pos;

  const int tr = wave >> 1;  // 0..3 -> tile row within region
  const int tc = wave & 1;   // 0..1 -> tile col within region

  v8f accv = {};
#if HAVE_TDM
  __shared__ float lA[32 * 64];
  __shared__ float lB[32 * 32];
  const unsigned ldsA = (unsigned)(unsigned long long)(uintptr_t)lA;
  const unsigned ldsB = (unsigned)(unsigned long long)(uintptr_t)lB;
  for (int k0 = 0; k0 < B; k0 += 32) {
    if (wave == 0) {
      tdm_load_2d(ldsA, xp + (long)k0 * rowStride + c0, 64, 32, rowStride);
      tdm_load_2d(ldsB, xp + (long)k0 * rowStride + d0, 32, 32, rowStride);
      __builtin_amdgcn_s_wait_tensorcnt(0);
    }
    __syncthreads();
    // LDS panels are [k][col]: element (i,k) at l[(k)*W + i] -> sAi=1, sAk=W
    accv = gram_tile(accv, lA, 1, 64, lB, 1, 32, 32,
                     tr * 16, tc * 16, lane, 64, 32, false);
    __syncthreads();
  }
#else
  accv = gram_tile(accv, xp, 1, rowStride, xp, 1, rowStride, B,
                   c0 + tr * 16, d0 + tc * 16, lane, D, D, true);
#endif

  const float invBm1 = 1.0f / (float)(B - 1);
  const int hi = lane >> 4;
  const int nc = d0 + tc * 16 + (lane & 15);
  const float mn = mp[nc];
  float sum = 0.0f;
#pragma unroll
  for (int r = 0; r < 8; ++r) {
    int mc = c0 + tr * 16 + r + 8 * hi;
    float cov = (accv[r] - (float)B * mp[mc] * mn) * invBm1;
    if (mc != nc) sum += cov * cov;
  }
  for (int off = 16; off; off >>= 1) sum += __shfl_xor(sum, off, 32);
  if (lane == 0) atomicAdd(acc, covScale * sum);
}

// Row squared-norms
__global__ void k_rownorm(const float* __restrict__ x, long rows, int C,
                          float* __restrict__ out) {
  long r = (long)blockIdx.x * blockDim.x + threadIdx.x;
  if (r >= rows) return;
  const float* p = x + r * (long)C;
  float s = 0.0f;
  for (int c = 0; c < C; ++c) { float v = p[c]; s += v * v; }
  out[r] = s;
}

// Feature-space cdist via WMMA
__global__ void k_cdist_wmma(const float* __restrict__ za, const float* __restrict__ zb,
                             const float* __restrict__ a2, const float* __restrict__ b2,
                             int N, int M, int C, float* __restrict__ dist) {
  const int lane = threadIdx.x & 31;
  const int wave = threadIdx.x >> 5;
  const int b = blockIdx.y;
  const int tN = (N + 15) >> 4, tM = (M + 15) >> 4;
  const int tile = blockIdx.x * (blockDim.x >> 5) + wave;
  if (tile >= tN * tM) return;  // wave-uniform
  const int n0 = (tile / tM) << 4;
  const int m0 = (tile % tM) << 4;
  const float* pa = za + (long)b * N * C;
  const float* pb = zb + (long)b * M * C;

  v8f g = {};
  g = gram_tile(g, pa, C, 1, pb, C, 1, C, n0, m0, lane, N, M, true);

  const int hi = lane >> 4;
  const int mcol = m0 + (lane & 15);
#pragma unroll
  for (int r = 0; r < 8; ++r) {
    int nrow = n0 + r + 8 * hi;
    if (nrow < N && mcol < M) {
      float d2 = a2[(long)b * N + nrow] + b2[(long)b * M + mcol] - 2.0f * g[r];
      dist[((long)b * N + nrow) * M + mcol] = sqrtf(fmaxf(d2, 0.0f));
    }
  }
}

// Location-space cdist (D=2)
__global__ void k_cdist_loc(const float* __restrict__ la, const float* __restrict__ lb,
                            int N, int M, long total, float* __restrict__ dist) {
  long idx = (long)blockIdx.x * blockDim.x + threadIdx.x;
  if (idx >= total) return;
  int m = (int)(idx % M);
  long t = idx / M;
  int n = (int)(t % N);
  int b = (int)(t / N);
  const float* ap = la + ((long)b * N + n) * 2;
  const float* bp = lb + ((long)b * M + m) * 2;
  float dx = ap[0] - bp[0], dy = ap[1] - bp[1];
  dist[idx] = sqrtf(fmaxf(dx * dx + dy * dy, 0.0f));
}

__global__ void k_rowmin(const float* __restrict__ dist, int N, int M,
                         float* __restrict__ nnd, int* __restrict__ nni) {
  long id = (long)blockIdx.x * blockDim.x + threadIdx.x;
  if (id >= (long)Bn * N) return;
  int n = (int)(id % N);
  int b = (int)(id / N);
  const float* p = dist + ((long)b * N + n) * M;
  float best = p[0]; int bi = 0;
  for (int m = 1; m < M; ++m) { float v = p[m]; if (v < best) { best = v; bi = m; } }
  nnd[id] = best; nni[id] = bi;
}

__global__ void k_colmin(const float* __restrict__ dist, int N, int M,
                         float* __restrict__ nnd, int* __restrict__ nni) {
  long id = (long)blockIdx.x * blockDim.x + threadIdx.x;
  if (id >= (long)Bn * M) return;
  int m = (int)(id % M);
  int b = (int)(id / M);
  const float* p = dist + (long)b * N * M + m;
  float best = p[0]; int bi = 0;
  for (int n = 1; n < N; ++n) { float v = p[(long)n * M]; if (v < best) { best = v; bi = n; } }
  nnd[id] = best; nni[id] = bi;
}

__global__ void k_topk(const float* __restrict__ nnd, int N, int K,
                       int* __restrict__ minIdx) {
  __shared__ float s[256];
  int b = blockIdx.x;
  for (int i = threadIdx.x; i < N; i += blockDim.x) s[i] = nnd[(long)b * N + i];
  __syncthreads();
  if (threadIdx.x == 0) {
    for (int t = 0; t < K; ++t) {
      float best = s[0]; int bi = 0;
      for (int i = 1; i < N; ++i) if (s[i] < best) { best = s[i]; bi = i; }
      minIdx[b * K + t] = bi;
      s[bi] = 3.402823466e38f;
    }
  }
}

__global__ void k_gather(const float* __restrict__ src, const float* __restrict__ cand,
                         const int* __restrict__ minIdx, const int* __restrict__ nnIdx,
                         int srcRows, int candRows, int C, int S, long total,
                         float* __restrict__ filt, float* __restrict__ nn) {
  long idx = (long)blockIdx.x * blockDim.x + threadIdx.x;
  if (idx >= total) return;
  int c = (int)(idx % C);
  long t = idx / C;
  int s = (int)(t % S);
  int b = (int)(t / S);
  int row = minIdx[b * S + s];
  filt[idx] = src[((long)b * srcRows + row) * C + c];
  int nrow = nnIdx[(long)b * srcRows + row];
  nn[idx] = cand[((long)b * candRows + nrow) * C + c];
}

// ---------------- host orchestration ----------------
static inline int cdiv(long a, int b) { return (int)((a + b - 1) / b); }

extern "C" void kernel_launch(void* const* d_in, const int* in_sizes, int n_in,
                              void* d_out, int out_size, void* d_ws, size_t ws_size,
                              hipStream_t stream) {
  const float* gf0    = (const float*)d_in[0];
  const float* gf1    = (const float*)d_in[1];
  const float* lf0    = (const float*)d_in[2];
  const float* lf1    = (const float*)d_in[3];
  const float* grid0  = (const float*)d_in[4];
  const float* grid1  = (const float*)d_in[5];
  const float* lgf0   = (const float*)d_in[6];
  const float* lgf1   = (const float*)d_in[7];
  const float* llf0   = (const float*)d_in[8];
  const float* llf1   = (const float*)d_in[9];
  const float* lgrid0 = (const float*)d_in[10];
  const float* lgrid1 = (const float*)d_in[11];

  float* ws = (float*)d_ws;
  size_t off = 0;
  float* ACC   = ws;                off += 16;
  float* MEANS = ws + off;          off += 16384;
  float* NRM0  = ws + off;          off += (size_t)Bn * NG;
  float* NRM1  = ws + off;          off += (size_t)Bn * NG;
  float* NRM2  = ws + off;          off += (size_t)Bn * NL;
  float* NRM3  = ws + off;          off += (size_t)Bn * NL;
  float* DIST  = ws + off;          off += (size_t)Bn * NG * NG;
  float* NND_A = ws + off;          off += (size_t)Bn * NG;
  int*   NNI_A = (int*)(ws + off);  off += (size_t)Bn * NG;
  float* NND_B = ws + off;          off += (size_t)Bn * NG;
  int*   NNI_B = (int*)(ws + off);  off += (size_t)Bn * NG;
  int*   MIDX_A = (int*)(ws + off); off += (size_t)Bn * KA;
  int*   MIDX_B = (int*)(ws + off); off += (size_t)Bn * KB;
  float* AFILT = ws + off;          off += (size_t)Bn * KA * DL;
  float* ANN   = ws + off;          off += (size_t)Bn * KA * DL;
  float* BFILT = ws + off;          off += (size_t)Bn * KB * DL;
  float* BNN   = ws + off;          off += (size_t)Bn * KB * DL;

  k_init<<<1, 32, 0, stream>>>(ACC);

  // ======== global loss (weight ALPHA) ========
  {
    long nG = (long)Bn * DG;
    float invS = ALPHA_ * LAMBDA_ / (6.0f * (float)nG);
    int blk = cdiv(nG, 256) < 512 ? cdiv(nG, 256) : 512;
    k_sumsq_diff<<<blk, 256, 0, stream>>>(gf0, gf1, nG, 2.0f * invS, ACC);
    k_sumsq_diff<<<blk, 256, 0, stream>>>(gf0, lgf0, nG, invS, ACC);
    k_sumsq_diff<<<blk, 256, 0, stream>>>(gf0, lgf1, nG, invS, ACC);
    k_sumsq_diff<<<blk, 256, 0, stream>>>(gf1, lgf0, nG, invS, ACC);
    k_sumsq_diff<<<blk, 256, 0, stream>>>(gf1, lgf1, nG, invS, ACC);

    float varS = ALPHA_ * MU_ / (4.0f * (float)DG);
    float covS = ALPHA_ * NU_ / (4.0f * (float)DG);
    const float* gmats[4] = {gf0, gf1, lgf0, lgf1};
    int covRegions = (DG / 64) * (DG / 32);  // 2048 blocks of 8 waves
    for (int t = 0; t < 4; ++t) {
      k_colstats<<<cdiv(DG, 256), 256, 0, stream>>>(gmats[t], Bn, DG, DG, MEANS, varS, ACC);
      k_cov_wmma<<<dim3(covRegions, 1), 256, 0, stream>>>(gmats[t], Bn, DG, DG, DG,
                                                          MEANS, covS, ACC);
    }
  }

  // ======== local loss (weight 1-ALPHA, averaged over 6 pairs) ========
  k_rownorm<<<cdiv((long)Bn * NG, 256), 256, 0, stream>>>(lf0, (long)Bn * NG, DL, NRM0);
  k_rownorm<<<cdiv((long)Bn * NG, 256), 256, 0, stream>>>(lf1, (long)Bn * NG, DL, NRM1);
  k_rownorm<<<cdiv((long)Bn * NL, 256), 256, 0, stream>>>(llf0, (long)Bn * NL, DL, NRM2);
  k_rownorm<<<cdiv((long)Bn * NL, 256), 256, 0, stream>>>(llf1, (long)Bn * NL, DL, NRM3);

  const float pairScale = (1.0f - ALPHA_) / 6.0f * 0.5f;

  auto vicreg = [&](const float* zfa, const float* zfb, int S) {
    long n = (long)Bn * S * DL;
    float invS = pairScale * LAMBDA_ / (float)n;
    int blk = cdiv(n, 256) < 512 ? cdiv(n, 256) : 512;
    k_sumsq_diff<<<blk, 256, 0, stream>>>(zfa, zfb, n, invS, ACC);
    float varS = pairScale * MU_ * 0.5f / (float)(S * DL);
    float covS = pairScale * (NU_ * 0.5f) / (float)(S * DL);
    const float* zz[2] = {zfa, zfb};
    int covRegions = (DL / 64) * (DL / 32);  // 128 blocks per position
    for (int t = 0; t < 2; ++t) {
      k_colstats<<<cdiv(S * DL, 256), 256, 0, stream>>>(zz[t], Bn, S * DL, (long)S * DL,
                                                        MEANS, varS, ACC);
      k_cov_wmma<<<dim3(covRegions, S), 256, 0, stream>>>(zz[t], Bn, DL, (long)S * DL,
                                                          DL, MEANS, covS, ACC);
    }
  };

  struct LM { const float* z; const float* g; int n; const float* nrm; };
  LM gl[2] = {{lf0, grid0, NG, NRM0}, {lf1, grid1, NG, NRM1}};
  LM ll[2] = {{llf0, lgrid0, NL, NRM2}, {llf1, lgrid1, NL, NRM3}};

  for (int i = 0; i < 2; ++i) {
    LM za = gl[i];
    LM others[3] = {gl[1 - i], ll[0], ll[1]};
    for (int o = 0; o < 3; ++o) {
      LM zb = others[o];
      const int N = za.n, M = zb.n;
      for (int dt = 0; dt < 2; ++dt) {
        if (dt == 0) {
          int tiles = ((N + 15) / 16) * ((M + 15) / 16);
          k_cdist_wmma<<<dim3(cdiv(tiles, 8), Bn), 256, 0, stream>>>(
              za.z, zb.z, za.nrm, zb.nrm, N, M, DL, DIST);
        } else {
          long tot = (long)Bn * N * M;
          k_cdist_loc<<<cdiv(tot, 256), 256, 0, stream>>>(za.g, zb.g, N, M, tot, DIST);
        }
        k_rowmin<<<cdiv((long)Bn * N, 256), 256, 0, stream>>>(DIST, N, M, NND_A, NNI_A);
        k_colmin<<<cdiv((long)Bn * M, 256), 256, 0, stream>>>(DIST, N, M, NND_B, NNI_B);
        k_topk<<<Bn, 64, 0, stream>>>(NND_A, N, KA, MIDX_A);
        k_topk<<<Bn, 64, 0, stream>>>(NND_B, M, KB, MIDX_B);
        long tA = (long)Bn * KA * DL;
        k_gather<<<cdiv(tA, 256), 256, 0, stream>>>(za.z, zb.z, MIDX_A, NNI_A,
                                                    N, M, DL, KA, tA, AFILT, ANN);
        long tB = (long)Bn * KB * DL;
        k_gather<<<cdiv(tB, 256), 256, 0, stream>>>(zb.z, za.z, MIDX_B, NNI_B,
                                                    M, N, DL, KB, tB, BFILT, BNN);
        vicreg(AFILT, ANN, KA);
        vicreg(BFILT, BNN, KB);
      }
    }
  }

  k_finalize<<<1, 1, 0, stream>>>(ACC, (float*)d_out);
  (void)in_sizes; (void)n_in; (void)out_size; (void)ws_size;
}
